// LSTM_Decoder_79697413144973
// MI455X (gfx1250) — compile-verified, matrix operand order checked by reading, not run on hardware
//
#include <hip/hip_runtime.h>
#include <hip/hip_bf16.h>

// Problem constants (reference: B,L,E,H,V,T,NL = 32,512,512,1024,32000,64,2)
#define B_ 32
#define L_ 512
#define E_ 512
#define H_ 1024
#define V_ 32000
#define T_ 64

typedef __attribute__((ext_vector_type(16))) __bf16 v16bf;
typedef __attribute__((ext_vector_type(8)))  float  v8f;

// Native f32 -> bf16 conversion; should lower to v_cvt_*bf16_f32 on gfx1250.
__device__ __forceinline__ __bf16 f2bf(float f) { return (__bf16)f; }

// Load one lane's 16 bf16 elements of a 16x32 WMMA operand tile.
// Per ISA 7.12.2 (16-bit A-matrix 16x32): lanes 0-15 hold row M=lane,
// K = {0..7, 16..23}; lanes 16-31 hold row M=lane-16, K = {8..15, 24..31}.
// B-matrix (32x16 = W^T for row-major W[N][K]) mirrors this with N=lane%16.
__device__ __forceinline__ v16bf load_tile(const float* __restrict__ base,
                                           int ld, int row, int k0, int lane) {
  const float* p = base + (size_t)row * ld + k0 + ((lane & 16) ? 8 : 0);
  float4 x0 = *(const float4*)(p);
  float4 x1 = *(const float4*)(p + 4);
  float4 x2 = *(const float4*)(p + 16);
  float4 x3 = *(const float4*)(p + 20);
  v16bf r;
  r[0]  = f2bf(x0.x); r[1]  = f2bf(x0.y); r[2]  = f2bf(x0.z); r[3]  = f2bf(x0.w);
  r[4]  = f2bf(x1.x); r[5]  = f2bf(x1.y); r[6]  = f2bf(x1.z); r[7]  = f2bf(x1.w);
  r[8]  = f2bf(x2.x); r[9]  = f2bf(x2.y); r[10] = f2bf(x2.z); r[11] = f2bf(x2.w);
  r[12] = f2bf(x3.x); r[13] = f2bf(x3.y); r[14] = f2bf(x3.z); r[15] = f2bf(x3.w);
  return r;
}

// Generic fused GEMM: C[M,N] = A1[M,K1] @ W1[N,K1]^T (+ A2[M,K2] @ W2[N,K2]^T)
//                              (+ b1[N]) (+ b2[N])
// One wave computes MT vertically-stacked 16x16 output tiles sharing a single
// B (weight) operand register set -> W global traffic is read once per N-tile
// instead of once per (M-tile, N-tile). MT=2 covers the whole M=32 batch.
// grid = (M/(16*MT), ceil(N/16/4)), block = 128 (4 waves, one N-tile each).
template <int MT>
__global__ __launch_bounds__(128)
void gemm2_wmma(const float* __restrict__ A1, int K1,
                const float* __restrict__ W1, int ldw1,
                const float* __restrict__ A2, int K2,
                const float* __restrict__ W2, int ldw2,
                const float* __restrict__ b1, const float* __restrict__ b2,
                float* __restrict__ C, long long ldc, int N) {
  const int lane   = threadIdx.x & 31;
  const int wave   = threadIdx.x >> 5;
  const int mtile0 = blockIdx.x * MT;
  const int ntile  = blockIdx.y * 4 + wave;
  if (ntile * 16 >= N) return;
  const int r15  = lane & 15;
  const int wrow = ntile * 16 + r15;

  v8f acc[MT];
#pragma unroll
  for (int m = 0; m < MT; ++m) acc[m] = (v8f){};

  for (int k0 = 0; k0 < K1; k0 += 32) {
    v16bf b = load_tile(W1, ldw1, wrow, k0, lane);
#pragma unroll
    for (int m = 0; m < MT; ++m) {
      v16bf a = load_tile(A1, K1, (mtile0 + m) * 16 + r15, k0, lane);
      acc[m] = __builtin_amdgcn_wmma_f32_16x16x32_bf16(false, a, false, b,
                                                       (short)0, acc[m],
                                                       false, false);
    }
  }
  for (int k0 = 0; k0 < K2; k0 += 32) {
    v16bf b = load_tile(W2, ldw2, wrow, k0, lane);
#pragma unroll
    for (int m = 0; m < MT; ++m) {
      v16bf a = load_tile(A2, K2, (mtile0 + m) * 16 + r15, k0, lane);
      acc[m] = __builtin_amdgcn_wmma_f32_16x16x32_bf16(false, a, false, b,
                                                       (short)0, acc[m],
                                                       false, false);
    }
  }

  const int n = ntile * 16 + r15;
  float bias = 0.f;
  if (b1) bias += b1[n];
  if (b2) bias += b2[n];
  // C/D layout (ISA 7.12.2): VGPR r -> M = r + 8*(lane>=16), N = lane%16.
#pragma unroll
  for (int m = 0; m < MT; ++m) {
    const int mbase = (mtile0 + m) * 16 + ((lane & 16) ? 8 : 0);
#pragma unroll
    for (int r = 0; r < 8; ++r)
      C[(long long)(mbase + r) * ldc + n] = acc[m][r] + bias;
  }
}

// scores[b,l] = sum_h tanh(enc_proj[b,l,h] + q[b,h]) * v_att[h]
// One wave per (b,l); wave32 shuffle reduction.
__global__ __launch_bounds__(256)
void attn_scores(const float* __restrict__ enc_proj, const float* __restrict__ q,
                 const float* __restrict__ v_att, float* __restrict__ scores) {
  const int lane = threadIdx.x & 31;
  const int wave = threadIdx.x >> 5;
  const int w = blockIdx.x * 8 + wave;
  const int b = w / L_, l = w % L_;
  const float* ep = enc_proj + ((size_t)b * L_ + l) * H_;
  const float* qb = q + b * H_;
  float s = 0.f;
  for (int h = lane; h < H_; h += 32)
    s += tanhf(ep[h] + qb[h]) * v_att[h];
#pragma unroll
  for (int off = 16; off > 0; off >>= 1)
    s += __shfl_xor(s, off, 32);
  if (lane == 0) scores[b * L_ + l] = s;
}

// softmax over L per batch, then ctx[b,h] = sum_l alpha[l] * enc_output[b,l,h]
__global__ __launch_bounds__(256)
void softmax_ctx(const float* __restrict__ scores,
                 const float* __restrict__ enc_output, float* __restrict__ ctx) {
  __shared__ float al[L_];
  __shared__ float red[256];
  const int b = blockIdx.x, tid = threadIdx.x;
  float m = -1e30f;
  for (int l = tid; l < L_; l += 256) {
    float v = scores[b * L_ + l]; al[l] = v; m = fmaxf(m, v);
  }
  red[tid] = m; __syncthreads();
  for (int s = 128; s > 0; s >>= 1) {
    if (tid < s) red[tid] = fmaxf(red[tid], red[tid + s]);
    __syncthreads();
  }
  m = red[0]; __syncthreads();
  float sum = 0.f;
  for (int l = tid; l < L_; l += 256) {
    float e = expf(al[l] - m); al[l] = e; sum += e;
  }
  red[tid] = sum; __syncthreads();
  for (int s = 128; s > 0; s >>= 1) {
    if (tid < s) red[tid] += red[tid + s];
    __syncthreads();
  }
  float inv = 1.f / red[0]; __syncthreads();
  for (int l = tid; l < L_; l += 256) al[l] *= inv;
  __syncthreads();
  for (int h = tid; h < H_; h += 256) {
    float acc = 0.f;
    const float* e = enc_output + (size_t)b * L_ * H_ + h;
    for (int l = 0; l < L_; ++l) acc += al[l] * e[(size_t)l * H_];
    ctx[b * H_ + h] = acc;
  }
}

// LSTM elementwise: gates g[B,4H] (torch order i,f,g,o) + old c -> new h,c (in place).
__global__ void lstm_ew(const float* __restrict__ g, float* __restrict__ h,
                        float* __restrict__ c) {
  int idx = blockIdx.x * blockDim.x + threadIdx.x;
  if (idx >= B_ * H_) return;
  int b = idx / H_, hh = idx % H_;
  const float* gb = g + (size_t)b * 4 * H_;
  float gi = gb[hh], gf = gb[H_ + hh], gg = gb[2 * H_ + hh], go = gb[3 * H_ + hh];
  float si = 1.f / (1.f + expf(-gi));
  float sf = 1.f / (1.f + expf(-gf));
  float so = 1.f / (1.f + expf(-go));
  float c2 = sf * c[idx] + si * tanhf(gg);
  c[idx] = c2;
  h[idx] = so * tanhf(c2);
}

// argmax over V per batch; writes next token (int) and _pred[b,t] (float).
__global__ __launch_bounds__(256)
void argmax_write(const float* __restrict__ pred, int* __restrict__ tok,
                  float* __restrict__ pred_out, int t) {
  __shared__ float bv[256];
  __shared__ int   bi[256];
  const int b = blockIdx.x, tid = threadIdx.x;
  const float* p = pred + (size_t)b * ((size_t)T_ * V_);
  float best = -1e30f; int bidx = 0;
  for (int v = tid; v < V_; v += 256) {
    float x = p[v];
    if (x > best) { best = x; bidx = v; }
  }
  bv[tid] = best; bi[tid] = bidx; __syncthreads();
  for (int s = 128; s > 0; s >>= 1) {
    if (tid < s) {
      if (bv[tid + s] > bv[tid] ||
          (bv[tid + s] == bv[tid] && bi[tid + s] < bi[tid])) {
        bv[tid] = bv[tid + s]; bi[tid] = bi[tid + s];
      }
    }
    __syncthreads();
  }
  if (tid == 0) { tok[b] = bi[0]; pred_out[b * T_ + t] = (float)bi[0]; }
}

__global__ void emb_gather(const float* __restrict__ emb, const int* __restrict__ tok,
                           float* __restrict__ et) {
  int b = blockIdx.x;
  int row = tok[b];
  for (int e = threadIdx.x; e < E_; e += blockDim.x)
    et[b * E_ + e] = emb[(size_t)row * E_ + e];
}

// Zero h/c, set tok from trg[:,0], zero outputs[:,0,:], _pred[:,0] = sos.
__global__ void init_kernel(const int* __restrict__ trg, const int* __restrict__ sos,
                            float* __restrict__ h, float* __restrict__ c,
                            int* __restrict__ tok, float* __restrict__ out0,
                            float* __restrict__ pred_out) {
  int idx = blockIdx.x * blockDim.x + threadIdx.x;
  const int total = B_ * V_;
  for (int i = idx; i < total; i += gridDim.x * blockDim.x)
    out0[(size_t)(i / V_) * T_ * V_ + (i % V_)] = 0.f;
  if (idx < 2 * B_ * H_) { h[idx] = 0.f; c[idx] = 0.f; }
  if (idx < B_) { tok[idx] = trg[idx * T_]; pred_out[idx * T_] = (float)sos[0]; }
}

extern "C" void kernel_launch(void* const* d_in, const int* in_sizes, int n_in,
                              void* d_out, int out_size, void* d_ws, size_t ws_size,
                              hipStream_t stream) {
  const float* enc_output = (const float*)d_in[0];
  const int*   trg        = (const int*)d_in[1];
  const float* emb        = (const float*)d_in[2];
  const float* W_att_enc  = (const float*)d_in[3];
  const float* b_att_enc  = (const float*)d_in[4];
  const float* W_att_dec  = (const float*)d_in[5];
  const float* b_att_dec  = (const float*)d_in[6];
  const float* v_att      = (const float*)d_in[7];
  const float* W_in       = (const float*)d_in[8];
  const float* b_in       = (const float*)d_in[9];
  const float* W_ih0      = (const float*)d_in[10];
  const float* W_hh0      = (const float*)d_in[11];
  const float* b_ih0      = (const float*)d_in[12];
  const float* b_hh0      = (const float*)d_in[13];
  const float* W_ih1      = (const float*)d_in[14];
  const float* W_hh1      = (const float*)d_in[15];
  const float* b_ih1      = (const float*)d_in[16];
  const float* b_hh1      = (const float*)d_in[17];
  const float* W_fc       = (const float*)d_in[18];
  const float* b_fc       = (const float*)d_in[19];
  const int*   sos_idx    = (const int*)d_in[23];

  float* outputs  = (float*)d_out;                       // [B,T,V]
  float* pred_tok = outputs + (size_t)B_ * T_ * V_;      // [B,T]

  // Workspace carve-out (~68 MB; enc_proj dominates).
  char* ws = (char*)d_ws;
  float* enc_proj = (float*)ws; ws += (size_t)B_ * L_ * H_ * 4;
  float* h   = (float*)ws; ws += (size_t)2 * B_ * H_ * 4;
  float* c   = (float*)ws; ws += (size_t)2 * B_ * H_ * 4;
  float* q   = (float*)ws; ws += (size_t)B_ * H_ * 4;
  float* sc  = (float*)ws; ws += (size_t)B_ * L_ * 4;
  float* ctx = (float*)ws; ws += (size_t)B_ * H_ * 4;
  float* et  = (float*)ws; ws += (size_t)B_ * E_ * 4;
  float* x   = (float*)ws; ws += (size_t)B_ * E_ * 4;
  float* g   = (float*)ws; ws += (size_t)B_ * 4 * H_ * 4;
  int*   tok = (int*)ws;   ws += (size_t)B_ * 4;

  float* h0 = h, *h1 = h + B_ * H_;
  float* c0 = c, *c1 = c + B_ * H_;

  // One-time: enc_proj = enc_output @ W_att_enc^T + b_att_enc  (16384x1024x1024)
  gemm2_wmma<2><<<dim3((B_ * L_) / 32, (H_ / 16) / 4), 128, 0, stream>>>(
      enc_output, H_, W_att_enc, H_, nullptr, 0, nullptr, 0,
      b_att_enc, nullptr, enc_proj, H_, H_);

  init_kernel<<<256, 256, 0, stream>>>(trg, sos_idx, h, c, tok, outputs, pred_tok);

  for (int t = 1; t < T_; ++t) {
    emb_gather<<<B_, 256, 0, stream>>>(emb, tok, et);

    // q = h1 @ W_att_dec^T + b_att_dec
    gemm2_wmma<2><<<dim3(1, 16), 128, 0, stream>>>(
        h1, H_, W_att_dec, H_, nullptr, 0, nullptr, 0,
        b_att_dec, nullptr, q, H_, H_);

    attn_scores<<<(B_ * L_) / 8, 256, 0, stream>>>(enc_proj, q, v_att, sc);
    softmax_ctx<<<B_, 256, 0, stream>>>(sc, enc_output, ctx);

    // x = emb_t @ W_in[:, :E]^T + ctx @ W_in[:, E:]^T + b_in
    gemm2_wmma<2><<<dim3(1, 8), 128, 0, stream>>>(
        et, E_, W_in, E_ + H_, ctx, H_, W_in + E_, E_ + H_,
        b_in, nullptr, x, E_, E_);

    // layer 0 gates + elementwise
    gemm2_wmma<2><<<dim3(1, 64), 128, 0, stream>>>(
        x, E_, W_ih0, E_, h0, H_, W_hh0, H_,
        b_ih0, b_hh0, g, 4 * H_, 4 * H_);
    lstm_ew<<<(B_ * H_ + 255) / 256, 256, 0, stream>>>(g, h0, c0);

    // layer 1 gates + elementwise
    gemm2_wmma<2><<<dim3(1, 64), 128, 0, stream>>>(
        h0, H_, W_ih1, H_, h1, H_, W_hh1, H_,
        b_ih1, b_hh1, g, 4 * H_, 4 * H_);
    lstm_ew<<<(B_ * H_ + 255) / 256, 256, 0, stream>>>(g, h1, c1);

    // pred written straight into d_out at [b, t, :] via ldc = T*V
    gemm2_wmma<2><<<dim3(1, 500), 128, 0, stream>>>(
        h1, H_, W_fc, H_, nullptr, 0, nullptr, 0,
        b_fc, nullptr, outputs + (size_t)t * V_, (long long)T_ * V_, V_);

    argmax_write<<<B_, 256, 0, stream>>>(outputs + (size_t)t * V_, tok, pred_tok, t);
  }
}
